// MinGRULayer_35459249996089
// MI455X (gfx1250) — matrix operand order repaired and checked by reading, not compiled, not run on hardware
//
#include <hip/hip_runtime.h>
#include <hip/hip_bf16.h>

// ---------------------------------------------------------------------------
// MinGRU layer for MI455X (gfx1250, wave32).
//   g/v/d = x @ W* + b*   (three 16384x1024x1024 GEMMs, bf16 WMMA, f32 accum)
//   x_scan = sigmoid(g)*tanh(v);  a = 0.001 + 0.998*sigmoid(d)
//   h_t = a_t h_{t-1} + x_scan_t  (chunked parallel scan, 32 chunks of 128)
// GEMM staging uses GLOBAL_LOAD_ASYNC_TO_LDS_B128 with explicit double
// buffering (ASYNCcnt) when the builtin is available; sync fallback otherwise.
// ---------------------------------------------------------------------------

typedef __attribute__((ext_vector_type(16))) __bf16 v16bf;
typedef __attribute__((ext_vector_type(8)))  float  v8f;
typedef int v4i __attribute__((vector_size(16)));

#define DMODEL 1024
#define BATCH  4
#define SEQ    4096
#define MTOT   (BATCH * SEQ)        // 16384 rows
#define NCHUNK 32
#define CHUNK  (SEQ / NCHUNK)       // 128

// GEMM tiling: block = 256 threads = 8 waves (2x4), tile 64x128, K-step 32.
#define BM 64
#define BN 128
#define BK 32
#define NSTAGE (DMODEL / BK)        // 32 K-steps
#define LDA (BK + 8)   // bf16 elems; row stride 80B (16B-aligned, bank-skewed)
#define LDB (BK + 8)

#if __has_builtin(__builtin_amdgcn_global_load_async_to_lds_b128)
#define HAVE_ASYNC_LDS 1
#else
#define HAVE_ASYNC_LDS 0
#endif

static __device__ __forceinline__ unsigned short f2bfu(float f) {
    union { float f; unsigned u; } a; a.f = f;
    unsigned r = a.u + 0x7FFFu + ((a.u >> 16) & 1u);   // round-to-nearest-even
    return (unsigned short)(r >> 16);
}

union Frag16 { uint4 u[2]; v16bf v; };

// 16-byte global -> LDS copy (async on CDNA5 when available)
static __device__ __forceinline__ void cp16_g2l(const __bf16* g, __bf16* l) {
#if HAVE_ASYNC_LDS
    const __attribute__((address_space(1))) void* gv =
        (const __attribute__((address_space(1))) void*)g;
    __attribute__((address_space(3))) void* lv =
        (__attribute__((address_space(3))) void*)l;
    __builtin_amdgcn_global_load_async_to_lds_b128(
        (__attribute__((address_space(1))) v4i*)gv,
        (__attribute__((address_space(3))) v4i*)lv, 0, 0);
#else
    *(uint4*)l = *(const uint4*)g;
#endif
}

template <int N>
static __device__ __forceinline__ void wait_async() {
#if HAVE_ASYNC_LDS
#if __has_builtin(__builtin_amdgcn_s_wait_asynccnt)
    __builtin_amdgcn_s_wait_asynccnt(N);
#else
    asm volatile("s_wait_asynccnt %0" ::"i"(N) : "memory");
#endif
#endif
}

// --- prep: x (f32) -> bf16, 4 elements/thread -------------------------------
__global__ __launch_bounds__(256) void cvt_x_bf16(const float* __restrict__ in,
                                                  unsigned short* __restrict__ out) {
    int i = (blockIdx.x * 256 + threadIdx.x) * 4;
    float4 v = *(const float4*)(in + i);
    uint2 p;
    p.x = (unsigned)f2bfu(v.x) | ((unsigned)f2bfu(v.y) << 16);
    p.y = (unsigned)f2bfu(v.z) | ((unsigned)f2bfu(v.w) << 16);
    *(uint2*)(out + i) = p;
}

// --- prep: W (k,n) f32 -> Wt (n,k) bf16, 64x64 tiles via LDS ----------------
__global__ __launch_bounds__(256) void transpose_w_bf16(const float* __restrict__ W,
                                                        unsigned short* __restrict__ Wt) {
    __shared__ float s[64][65];
    const int k0 = blockIdx.y * 64, n0 = blockIdx.x * 64;
#pragma unroll
    for (int i = 0; i < 16; ++i) {
        int idx = threadIdx.x + i * 256;
        int kk = idx >> 6, nn = idx & 63;
        s[kk][nn] = W[(size_t)(k0 + kk) * DMODEL + n0 + nn];
    }
    __syncthreads();
#pragma unroll
    for (int i = 0; i < 16; ++i) {
        int idx = threadIdx.x + i * 256;
        int nn = idx >> 6, kk = idx & 63;
        Wt[(size_t)(n0 + nn) * DMODEL + k0 + kk] = f2bfu(s[kk][nn]);
    }
}

// --- fused triple-GEMM + bias + activation ----------------------------------
__global__ __launch_bounds__(256)
void gemm3_act(const __bf16* __restrict__ xb,
               const __bf16* __restrict__ Wtg,
               const __bf16* __restrict__ Wtv,
               const __bf16* __restrict__ Wtd,
               const float* __restrict__ bg,
               const float* __restrict__ bv,
               const float* __restrict__ bd,
               float* __restrict__ xs,    // sigmoid(g)*tanh(v)
               float* __restrict__ av)    // 0.001 + 0.998*sigmoid(d)
{
    __shared__ __align__(16) __bf16 lA[2][BM * LDA];        // double-buffered
    __shared__ __align__(16) __bf16 lB[2][3][BN * LDB];

    const int tid  = threadIdx.x;
    const int m0   = blockIdx.y * BM;
    const int n0   = blockIdx.x * BN;
    const int wave = tid >> 5, lane = tid & 31;
    const int wm = wave >> 2, wn = wave & 3;     // 2x4 wave grid
    const int lh = lane >> 4, ll = lane & 15;    // half-wave, lane-in-half

    // staging assignments: A = 1x16B/thread, B = 2x16B/thread per matrix
    const int a_row = tid >> 2, a_col = (tid & 3) * 8;
    const int b_n   = tid >> 1, b_k   = (tid & 1) * 16;

    const __bf16* Wts[3] = { Wtg, Wtv, Wtd };

    v8f acc[3][2][2];
    v8f zacc = {0.f, 0.f, 0.f, 0.f, 0.f, 0.f, 0.f, 0.f};
#pragma unroll
    for (int q = 0; q < 3; ++q)
#pragma unroll
        for (int i = 0; i < 2; ++i)
#pragma unroll
            for (int j = 0; j < 2; ++j) acc[q][i][j] = zacc;

    auto issue_stage = [&](int buf, int k0) {
        cp16_g2l(xb + (size_t)(m0 + a_row) * DMODEL + k0 + a_col,
                 &lA[buf][a_row * LDA + a_col]);
#pragma unroll
        for (int q = 0; q < 3; ++q) {
            const __bf16* w = Wts[q] + (size_t)(n0 + b_n) * DMODEL + k0 + b_k;
            cp16_g2l(w,     &lB[buf][q][b_n * LDB + b_k]);
            cp16_g2l(w + 8, &lB[buf][q][b_n * LDB + b_k + 8]);
        }
    };

    issue_stage(0, 0);

    for (int s = 0; s < NSTAGE; ++s) {
        const int buf = s & 1;
        if (s + 1 < NSTAGE) {
            issue_stage(buf ^ 1, (s + 1) * BK);   // prefetch next stage
            wait_async<7>();                      // stage s resident (in-order)
        } else {
            wait_async<0>();
        }
        __syncthreads();

        // A fragments: lane=m (0..15), K pairs [8h..8h+7] and [16+8h..23+8h]
        Frag16 afr[2];
#pragma unroll
        for (int fi = 0; fi < 2; ++fi) {
            const __bf16* p = &lA[buf][(wm * 32 + fi * 16 + ll) * LDA];
            afr[fi].u[0] = *(const uint4*)(p + 8 * lh);
            afr[fi].u[1] = *(const uint4*)(p + 16 + 8 * lh);
        }
#pragma unroll
        for (int q = 0; q < 3; ++q)
#pragma unroll
            for (int fj = 0; fj < 2; ++fj) {
                // B fragment: lane=n, K pairs [8h..8h+7] and [16+8h..23+8h]
                const __bf16* p = &lB[buf][q][(wn * 32 + fj * 16 + ll) * LDB];
                Frag16 bfr;
                bfr.u[0] = *(const uint4*)(p + 8 * lh);
                bfr.u[1] = *(const uint4*)(p + 16 + 8 * lh);
#pragma unroll
                for (int fi = 0; fi < 2; ++fi)
                    acc[q][fi][fj] = __builtin_amdgcn_wmma_f32_16x16x32_bf16(
                        false, afr[fi].v, false, bfr.v,
                        (short)0, acc[q][fi][fj], false, false);
            }
        __syncthreads();   // all reads of buf done before it is refilled
    }

    // epilogue: bias + activations, write x_scan and a
#pragma unroll
    for (int fj = 0; fj < 2; ++fj) {
        const int n = n0 + wn * 32 + fj * 16 + ll;
        const float vbg = bg[n], vbv = bv[n], vbd = bd[n];
#pragma unroll
        for (int fi = 0; fi < 2; ++fi) {
#pragma unroll
            for (int r = 0; r < 8; ++r) {
                const int m = m0 + wm * 32 + fi * 16 + lh * 8 + r;
                const float gg = acc[0][fi][fj][r] + vbg;
                const float vv = acc[1][fi][fj][r] + vbv;
                const float dd = acc[2][fi][fj][r] + vbd;
                const float sg = 1.0f / (1.0f + __expf(-gg));
                const float sd = 1.0f / (1.0f + __expf(-dd));
                const float e2 = __expf(-2.0f * fabsf(vv));
                const float th = copysignf((1.0f - e2) / (1.0f + e2), vv);
                const size_t o = (size_t)m * DMODEL + n;
                xs[o] = sg * th;
                av[o] = 0.001f + 0.998f * sd;
            }
        }
    }
}

// --- scan phase A: per-chunk (prod a, local h_end), no full h writes --------
__global__ __launch_bounds__(256)
void scan_phaseA(const float* __restrict__ a, const float* __restrict__ xs,
                 float* __restrict__ Atot, float* __restrict__ hend) {
    const int dd = blockIdx.x * 256 + threadIdx.x;     // channel
    const int b = blockIdx.y, c = blockIdx.z;
    size_t base = ((size_t)(b * SEQ + c * CHUNK)) * DMODEL + dd;
    float A = 1.0f, h = 0.0f;
    for (int t = 0; t < CHUNK; ++t) {
        size_t i = base + (size_t)t * DMODEL;
        float at = a[i];
        h = at * h + xs[i];
        A *= at;
    }
    const int ci = (b * NCHUNK + c) * DMODEL + dd;
    Atot[ci] = A;
    hend[ci] = h;
}

// --- scan phase B: sequential scan over 32 chunk carries per channel --------
__global__ __launch_bounds__(256)
void scan_phaseB(const float* __restrict__ Atot, const float* __restrict__ hend,
                 float* __restrict__ carry) {
    const int id = blockIdx.x * 256 + threadIdx.x;     // 0..4095
    const int b = id >> 10, dd = id & (DMODEL - 1);
    float run = 0.0f;
    for (int c = 0; c < NCHUNK; ++c) {
        const int ci = (b * NCHUNK + c) * DMODEL + dd;
        carry[ci] = run;                 // h entering chunk c
        run = Atot[ci] * run + hend[ci];
    }
}

// --- scan phase C: replay each chunk seeded with its carry, write h ---------
__global__ __launch_bounds__(256)
void scan_phaseC(const float* __restrict__ a, const float* __restrict__ xs,
                 const float* __restrict__ carry, float* __restrict__ out) {
    const int dd = blockIdx.x * 256 + threadIdx.x;
    const int b = blockIdx.y, c = blockIdx.z;
    size_t base = ((size_t)(b * SEQ + c * CHUNK)) * DMODEL + dd;
    float h = carry[(b * NCHUNK + c) * DMODEL + dd];
    for (int t = 0; t < CHUNK; ++t) {
        size_t i = base + (size_t)t * DMODEL;
        h = a[i] * h + xs[i];
        out[i] = h;
    }
}

// ---------------------------------------------------------------------------
extern "C" void kernel_launch(void* const* d_in, const int* in_sizes, int n_in,
                              void* d_out, int out_size, void* d_ws, size_t ws_size,
                              hipStream_t stream) {
    const float* x  = (const float*)d_in[0];
    const float* Wg = (const float*)d_in[1];
    const float* bg = (const float*)d_in[2];
    const float* Wv = (const float*)d_in[3];
    const float* bv = (const float*)d_in[4];
    const float* Wd = (const float*)d_in[5];
    const float* bd = (const float*)d_in[6];
    float* out = (float*)d_out;

    const size_t NTOK = (size_t)MTOT * DMODEL;     // 16,777,216
    const size_t WSZ  = (size_t)DMODEL * DMODEL;   // 1,048,576
    const size_t CSZ  = (size_t)BATCH * NCHUNK * DMODEL;

    char* w = (char*)d_ws;
    size_t off = 0;
    unsigned short* xb  = (unsigned short*)(w + off); off += NTOK * 2;
    unsigned short* wtg = (unsigned short*)(w + off); off += WSZ * 2;
    unsigned short* wtv = (unsigned short*)(w + off); off += WSZ * 2;
    unsigned short* wtd = (unsigned short*)(w + off); off += WSZ * 2;
    float* xs    = (float*)(w + off); off += NTOK * 4;
    float* av    = (float*)(w + off); off += NTOK * 4;
    float* Atot  = (float*)(w + off); off += CSZ * 4;
    float* hend  = (float*)(w + off); off += CSZ * 4;
    float* carry = (float*)(w + off); off += CSZ * 4;

    // 1) precision staging
    cvt_x_bf16<<<NTOK / (256 * 4), 256, 0, stream>>>(x, xb);
    dim3 tg(DMODEL / 64, DMODEL / 64);
    transpose_w_bf16<<<tg, 256, 0, stream>>>(Wg, wtg);
    transpose_w_bf16<<<tg, 256, 0, stream>>>(Wv, wtv);
    transpose_w_bf16<<<tg, 256, 0, stream>>>(Wd, wtd);

    // 2) fused triple GEMM + activations
    dim3 gg(DMODEL / BN, MTOT / BM);   // (8, 256)
    gemm3_act<<<gg, 256, 0, stream>>>((const __bf16*)xb, (const __bf16*)wtg,
                                      (const __bf16*)wtv, (const __bf16*)wtd,
                                      bg, bv, bd, xs, av);

    // 3) chunked linear scan
    dim3 gs(DMODEL / 256, BATCH, NCHUNK);   // (4, 4, 32)
    scan_phaseA<<<gs, 256, 0, stream>>>(av, xs, Atot, hend);
    scan_phaseB<<<(BATCH * DMODEL) / 256, 256, 0, stream>>>(Atot, hend, carry);
    scan_phaseC<<<gs, 256, 0, stream>>>(av, xs, carry, out);
}
